// MultiheadAttention_52278341927166
// MI455X (gfx1250) — matrix operand order repaired and checked by reading, not compiled, not run on hardware
//
#include <hip/hip_runtime.h>
#include <hip/hip_bf16.h>
#include <math.h>

// ---------------------------------------------------------------------------
// CDNA5 (gfx1250) WMMA bf16 tiled GEMM + row-wise glue kernels implementing:
//   qkv proj -> adj=x x^T -> ctx proj -> L2 norm -> fused scores (2 GEMMs with
//   scale/accumulate epilogue) -> softmax/mask/L1 -> attn@v -> output proj.
// v4: BTRANS is a template parameter (no uniform branch / SALU select chain in
//     the hot loop). Register-staged software pipeline: next tile's
//     global_load_b128s issue before the current tile's WMMAs and are drained
//     afterwards; one s_wait_loadcnt batch + one barrier per K-step.
// All GEMM dims are multiples of (128,64,32): no edge handling, EXEC all-ones
// around every v_wmma (ISA requirement).
// ---------------------------------------------------------------------------

typedef __attribute__((ext_vector_type(16))) __bf16 v16bf;
typedef __attribute__((ext_vector_type(8)))  float  v8f;

#define TILE_M 128
#define TILE_N 64
#define TILE_K 32
#define ASTR   40   // bf16 elements per LDS A row (32 + pad)
#define BSTR   40   // bf16 elements per LDS B row (32 + pad)

union Pk4 { __bf16 h[4]; uint2 u2; };

struct StageRegs { float4 a[4]; float4 b[2]; };

// Phase 1: issue global loads for one K-tile into registers (no waits here).
template <int BTRANS>
__device__ __forceinline__ void issue_loads(const float* __restrict__ Ab,
                                            const float* __restrict__ Bb,
                                            unsigned lda, unsigned ldb,
                                            unsigned row0, unsigned col0,
                                            unsigned k0, int tid, StageRegs& rg)
{
    #pragma unroll
    for (int t = 0; t < 4; ++t) {
        int e  = tid + t * 256;
        int r  = e >> 3;
        int c4 = (e & 7) << 2;
        unsigned off = (row0 + (unsigned)r) * lda + k0 + (unsigned)c4;
        rg.a[t] = *(const float4*)(Ab + off);
    }
    if (BTRANS) {
        #pragma unroll
        for (int t = 0; t < 2; ++t) {
            int e  = tid + t * 256;
            int n  = e >> 3;
            int c4 = (e & 7) << 2;
            unsigned off = (col0 + (unsigned)n) * ldb + k0 + (unsigned)c4;
            rg.b[t] = *(const float4*)(Bb + off);
        }
    } else {
        #pragma unroll
        for (int t = 0; t < 2; ++t) {
            int e  = tid + t * 256;
            int k  = e >> 4;                 // 16 float4 per 64-wide row
            int n4 = (e & 15) << 2;
            unsigned off = (k0 + (unsigned)k) * ldb + col0 + (unsigned)n4;
            rg.b[t] = *(const float4*)(Bb + off);
        }
    }
}

// Phase 2: convert f32->bf16 and store the staged registers into LDS.
template <int BTRANS>
__device__ __forceinline__ void store_lds(const StageRegs& rg,
                                          __bf16* __restrict__ As,
                                          __bf16* __restrict__ Bs, int tid)
{
    #pragma unroll
    for (int t = 0; t < 4; ++t) {
        int e  = tid + t * 256;
        int r  = e >> 3;
        int c4 = (e & 7) << 2;
        Pk4 pk;
        pk.h[0] = (__bf16)rg.a[t].x; pk.h[1] = (__bf16)rg.a[t].y;
        pk.h[2] = (__bf16)rg.a[t].z; pk.h[3] = (__bf16)rg.a[t].w;
        *(uint2*)(As + r * ASTR + c4) = pk.u2;          // ds_store_b64
    }
    if (BTRANS) {
        #pragma unroll
        for (int t = 0; t < 2; ++t) {
            int e  = tid + t * 256;
            int n  = e >> 3;
            int c4 = (e & 7) << 2;
            Pk4 pk;
            pk.h[0] = (__bf16)rg.b[t].x; pk.h[1] = (__bf16)rg.b[t].y;
            pk.h[2] = (__bf16)rg.b[t].z; pk.h[3] = (__bf16)rg.b[t].w;
            *(uint2*)(Bs + n * BSTR + c4) = pk.u2;
        }
    } else {
        #pragma unroll
        for (int t = 0; t < 2; ++t) {
            int e  = tid + t * 256;
            int k  = e >> 4;
            int n4 = (e & 15) << 2;
            Bs[(n4 + 0) * BSTR + k] = (__bf16)rg.b[t].x;
            Bs[(n4 + 1) * BSTR + k] = (__bf16)rg.b[t].y;
            Bs[(n4 + 2) * BSTR + k] = (__bf16)rg.b[t].z;
            Bs[(n4 + 3) * BSTR + k] = (__bf16)rg.b[t].w;
        }
    }
}

// C[b] = alpha * A[b] (MxK) * op(B[b]) (KxN) [+ bias[n]] [+ c0] [+ C[b] if acc]
// BTRANS=1: B stored as [N][K] (we compute A*B^T); BTRANS=0: B stored [K][N].
template <int BTRANS>
__global__ __launch_bounds__(256)
void gemm_bf16_wmma(const float* __restrict__ A, unsigned lda, long long strideA,
                    const float* __restrict__ B, unsigned ldb, long long strideB,
                    float* __restrict__ C, unsigned ldc, long long strideC,
                    int K,
                    const float* __restrict__ alpha_ptr,
                    const float* __restrict__ bias_ptr,
                    const float* __restrict__ c0_ptr,
                    int accumulate)
{
    __shared__ __bf16 As[2][TILE_M * ASTR];
    __shared__ __bf16 Bs[2][TILE_N * BSTR];

    const int tid  = threadIdx.x;
    const int lane = tid & 31;          // wave32
    const int wid  = tid >> 5;          // 8 waves / block
    const int waveM = wid >> 1;         // 0..3 -> 32-row strip
    const int waveN = wid & 1;          // 0..1 -> 32-col strip
    const int lhi  = lane >> 4;         // lane half (ISA fragment layout)
    const int l15  = lane & 15;

    const unsigned row0 = blockIdx.y * TILE_M;
    const unsigned col0 = blockIdx.x * TILE_N;

    const float* Ab = A + (long long)blockIdx.z * strideA;
    const float* Bb = B + (long long)blockIdx.z * strideB;
    float*       Cb = C + (long long)blockIdx.z * strideC;

    v8f acc[2][2] = {};
    StageRegs rg;

    // Prologue: stage tile 0.
    issue_loads<BTRANS>(Ab, Bb, lda, ldb, row0, col0, 0, tid, rg);
    store_lds<BTRANS>(rg, As[0], Bs[0], tid);
    __syncthreads();

    const int nk = K / TILE_K;
    for (int kt = 0; kt < nk; ++kt) {
        const int cur = kt & 1;
        const bool more = (kt + 1 < nk);

        // Issue next tile's global loads BEFORE the matrix work (in flight
        // underneath the ds_load + wmma phase; drained in store_lds below).
        if (more)
            issue_loads<BTRANS>(Ab, Bb, lda, ldb, row0, col0,
                                (unsigned)(kt + 1) * TILE_K, tid, rg);

        const unsigned int* Asu = reinterpret_cast<const unsigned int*>(As[cur]);
        const unsigned int* Bsu = reinterpret_cast<const unsigned int*>(Bs[cur]);

        // ---- gather fragments per ISA 16-bit A(16x32)/B(32x16) layouts --
        union FragA { unsigned int u[8]; v16bf v; } a[2];
        union FragB { unsigned int u[8]; v16bf v; } bfr[2];
        #pragma unroll
        for (int mt = 0; mt < 2; ++mt) {
            int m = waveM * 32 + mt * 16 + l15;
            int base = m * (ASTR / 2);
            #pragma unroll
            for (int p = 0; p < 8; ++p) {
                // k-pair: p<4 -> 2p + 8*lhi ; p>=4 -> 16 + 2(p-4) + 8*lhi
                int ku = (p < 4 ? p : p + 4) + 4 * lhi;   // in uint units
                a[mt].u[p] = Asu[base + ku];
            }
        }
        #pragma unroll
        for (int nt = 0; nt < 2; ++nt) {
            int n = waveN * 32 + nt * 16 + l15;
            int base = n * (BSTR / 2);
            #pragma unroll
            for (int p = 0; p < 8; ++p) {
                // k-pair: p<4 -> 2p + 8*lhi ; p>=4 -> 16 + 2(p-4) + 8*lhi
                bfr[nt].u[p] = Bsu[base + (p < 4 ? p : p + 4) + 4 * lhi];
            }
        }
        // ---- 4x v_wmma_f32_16x16x32_bf16 --------------------------------
        #pragma unroll
        for (int mt = 0; mt < 2; ++mt)
            #pragma unroll
            for (int nt = 0; nt < 2; ++nt)
                acc[mt][nt] = __builtin_amdgcn_wmma_f32_16x16x32_bf16(
                    false, a[mt].v, false, bfr[nt].v,
                    (short)0, acc[mt][nt], false, false);

        // Drain the staged loads into the other LDS buffer, then one barrier.
        if (more)
            store_lds<BTRANS>(rg, As[cur ^ 1], Bs[cur ^ 1], tid);
        __syncthreads();
    }

    const float alpha = alpha_ptr ? alpha_ptr[0] : 1.0f;
    const float c0    = c0_ptr    ? c0_ptr[0]    : 0.0f;

    // C/D layout: VGPR r -> M = r + 8*lhi ; N = l15
    #pragma unroll
    for (int mt = 0; mt < 2; ++mt) {
        #pragma unroll
        for (int nt = 0; nt < 2; ++nt) {
            unsigned gn = col0 + waveN * 32 + nt * 16 + l15;
            float bias = bias_ptr ? bias_ptr[gn] : 0.0f;
            #pragma unroll
            for (int r = 0; r < 8; ++r) {
                unsigned gm = row0 + waveM * 32 + mt * 16 + lhi * 8 + r;
                unsigned idx = gm * ldc + gn;
                float v = alpha * acc[mt][nt][r] + c0 + bias;
                if (accumulate) v += Cb[idx];
                Cb[idx] = v;
            }
        }
    }
}

// L2-normalize rows of length `rowlen` packed 2 per `ld` row (q-half/k-half).
__global__ __launch_bounds__(256)
void l2norm_rows(float* __restrict__ data, long long ld, int rowlen, float eps)
{
    __shared__ float red[256];
    long long base = (long long)(blockIdx.x >> 1) * ld
                   + (long long)(blockIdx.x & 1) * rowlen;
    const int tid = threadIdx.x;
    float s = 0.f;
    for (int i = tid; i < rowlen; i += 256) { float v = data[base + i]; s += v * v; }
    red[tid] = s; __syncthreads();
    for (int o = 128; o > 0; o >>= 1) {
        if (tid < o) red[tid] += red[tid + o];
        __syncthreads();
    }
    float inv = 1.0f / fmaxf(sqrtf(red[0]), eps);
    __syncthreads();
    for (int i = tid; i < rowlen; i += 256) data[base + i] *= inv;
}

// softmax over row, then post-softmax mask, then L1 renorm (eps-guarded).
__global__ __launch_bounds__(256)
void softmax_mask_l1(float* __restrict__ scores, const int* __restrict__ mask,
                     int S, float eps)
{
    __shared__ float red[256];
    long long base = (long long)blockIdx.x * S;
    const int tid = threadIdx.x;

    float mx = -INFINITY;
    for (int i = tid; i < S; i += 256) mx = fmaxf(mx, scores[base + i]);
    red[tid] = mx; __syncthreads();
    for (int o = 128; o > 0; o >>= 1) {
        if (tid < o) red[tid] = fmaxf(red[tid], red[tid + o]);
        __syncthreads();
    }
    mx = red[0]; __syncthreads();

    float s = 0.f;
    for (int i = tid; i < S; i += 256) {
        float e = __expf(scores[base + i] - mx);
        scores[base + i] = e; s += e;
    }
    red[tid] = s; __syncthreads();
    for (int o = 128; o > 0; o >>= 1) {
        if (tid < o) red[tid] += red[tid + o];
        __syncthreads();
    }
    float invsum = 1.0f / red[0]; __syncthreads();

    float ms = 0.f;
    for (int i = tid; i < S; i += 256) {
        float p = scores[base + i] * invsum;
        p = mask[base + i] ? p : 0.0f;
        scores[base + i] = p; ms += p;
    }
    red[tid] = ms; __syncthreads();
    for (int o = 128; o > 0; o >>= 1) {
        if (tid < o) red[tid] += red[tid + o];
        __syncthreads();
    }
    float invl1 = 1.0f / fmaxf(red[0], eps);
    __syncthreads();
    for (int i = tid; i < S; i += 256) scores[base + i] *= invl1;
}

extern "C" void kernel_launch(void* const* d_in, const int* in_sizes, int n_in,
                              void* d_out, int out_size, void* d_ws, size_t ws_size,
                              hipStream_t stream)
{
    (void)in_sizes; (void)n_in; (void)out_size; (void)ws_size;
    const float* x    = (const float*)d_in[0];
    const int*   mask = (const int*)  d_in[1];
    const float* Wqkv = (const float*)d_in[2];
    const float* bqkv = (const float*)d_in[3];
    const float* Wctx = (const float*)d_in[4];
    const float* bctx = (const float*)d_in[5];
    const float* Wf   = (const float*)d_in[6];
    const float* bf   = (const float*)d_in[7];
    const float* Wo   = (const float*)d_in[8];
    const float* bo   = (const float*)d_in[9];
    float* out = (float*)d_out;

    const int B = 16, S = 1024;
    const long long SS  = (long long)S * S;          // 1M
    const long long SD  = (long long)S * 1024;       // 1M
    const long long S2E = (long long)S * 2048;       // 2M
    const float EPS = 1e-12f;

    float* ws     = (float*)d_ws;
    float* qkv    = ws;                              // (B*S, 2048)  128MB
    float* adj    = ws + 33554432ll;                 // (B, S, S)     64MB
    float* ctx    = ws + 50331648ll;                 // (B*S, 2048)  128MB
    float* sc     = ws + 83886080ll;                 // (B, S, S)     64MB
    float* values = qkv;                             // reuse after q/k consumed

    dim3 blk(256);

    // 1) qkv = x * Wqkv^T + bqkv        (16384 x 2048 x 1024)
    gemm_bf16_wmma<1><<<dim3(2048/TILE_N, 16384/TILE_M, 1), blk, 0, stream>>>(
        x, 1024, 0, Wqkv, 1024, 0, qkv, 2048, 0,
        1024, nullptr, bqkv, nullptr, 0);

    // 2) adj = x * x^T per batch        (16 x 1024^3)
    gemm_bf16_wmma<1><<<dim3(1024/TILE_N, 1024/TILE_M, B), blk, 0, stream>>>(
        x, 1024, SD, x, 1024, SD, adj, 1024, SS,
        1024, nullptr, nullptr, nullptr, 0);

    // 3) ctx = adj * Wctx^T + bctx      (16 x 1024 x 2048 x 1024)
    gemm_bf16_wmma<1><<<dim3(2048/TILE_N, 1024/TILE_M, B), blk, 0, stream>>>(
        adj, 1024, SS, Wctx, 1024, 0, ctx, 2048, S2E,
        1024, nullptr, bctx, nullptr, 0);

    // 4) L2-normalize ctx_q / ctx_k rows
    l2norm_rows<<<B * S * 2, blk, 0, stream>>>(ctx, 2048, 1024, EPS);

    // 5) sc = Wf[0] * (q k^T) + bf
    gemm_bf16_wmma<1><<<dim3(1024/TILE_N, 1024/TILE_M, B), blk, 0, stream>>>(
        qkv, 2048, S2E, qkv + 1024, 2048, S2E, sc, 1024, SS,
        1024, Wf + 0, nullptr, bf, 0);

    // 6) sc += Wf[1] * (ctx_q ctx_k^T)
    gemm_bf16_wmma<1><<<dim3(1024/TILE_N, 1024/TILE_M, B), blk, 0, stream>>>(
        ctx, 2048, S2E, ctx + 1024, 2048, S2E, sc, 1024, SS,
        1024, Wf + 1, nullptr, nullptr, 1);

    // 7) softmax -> mask -> L1 renorm (in place)
    softmax_mask_l1<<<B * S, blk, 0, stream>>>(sc, mask, S, EPS);

    // 8) values = attn * x              (B normal-orientation)
    gemm_bf16_wmma<0><<<dim3(1024/TILE_N, 1024/TILE_M, B), blk, 0, stream>>>(
        sc, 1024, SS, x, 1024, SD, values, 1024, SD,
        1024, nullptr, nullptr, nullptr, 0);

    // 9) out = x * Wo[:, :D]^T + bo ; out += values * Wo[:, D:]^T
    gemm_bf16_wmma<1><<<dim3(1024/TILE_N, 16384/TILE_M, 1), blk, 0, stream>>>(
        x, 1024, 0, Wo, 2048, 0, out, 1024, 0,
        1024, nullptr, bo, nullptr, 0);
    gemm_bf16_wmma<1><<<dim3(1024/TILE_N, 16384/TILE_M, 1), blk, 0, stream>>>(
        values, 1024, 0, Wo + 1024, 2048, 0, out, 1024, 0,
        1024, nullptr, nullptr, nullptr, 1);
}
